// OutputBlock_45148696215937
// MI455X (gfx1250) — matrix operand order repaired and checked by reading, not compiled, not run on hardware
//
#include <hip/hip_runtime.h>
#include <hip/hip_bf16.h>

// ---------------------------------------------------------------------------
// MI455X (gfx1250, wave32) GemNet-style output block.
// Memory-bound: streaming x (512 MB) @ 23.3 TB/s ~= 26 us. Both the x@Wx GEMM
// and the rbf gate GEMM ride v_wmma_f32_16x16x32_f16 (gate K zero-padded to
// 32); x tiles are DMA'd into LDS by the Tensor Data Mover with hardware row
// padding; segment-sum via global_atomic_add_f32 into L2-resident d_ws.
// ---------------------------------------------------------------------------

typedef __attribute__((ext_vector_type(16))) _Float16     v16h;
typedef __attribute__((ext_vector_type(8)))  float        v8f;
typedef __attribute__((ext_vector_type(4)))  unsigned int u32x4;
typedef __attribute__((ext_vector_type(8)))  unsigned int u32x8;

// fast SiLU: v * rcp(1+exp(-v))  (v_exp_f32 + v_rcp_f32, no IEEE divide)
__device__ __forceinline__ float silu_f(float v) {
    return v * __builtin_amdgcn_rcpf(1.0f + __expf(-v));
}

#define EMB   128
#define NRBF  16
#define TE    128          // rows per block
#define XP    132          // padded f32 x-tile row stride (DWORDs) == TDM pad
#define KP    136          // padded f16 row stride for W^T
#define RP    40           // padded f16 row stride for Wrbf^T (K=32 + pad)

#define SMEM_EDGE (TE*XP*4 + EMB*KP*2 + EMB*RP*2 + TE*17*4 + EMB*4 + EMB*4 + TE*4)
#define SMEM_ATOM (TE*XP*4 + EMB*KP*2 + EMB*4)

// ---------------------------------------------------------------------------
// TDM: DMA a [rows<=128] x 128-f32 row-major tile from global into LDS at
// byte offset lds_off, inserting 4 DWORDs of padding after every 128 DWORDs
// (-> 132-DWORD LDS rows).  D# per CDNA5 ISA §8.3/§8.4; groups 2/3 zeroed
// (tile_dim2/3 = 0).  OOB rows (beyond rows_avail) read as zero.  Issue from
// ONE wave; EXEC is ignored by tensor ops.
// ---------------------------------------------------------------------------
__device__ __forceinline__ void tdm_load_tile_f32(const float* gptr,
                                                  unsigned lds_off,
                                                  unsigned rows_avail)
{
    unsigned long long ga = (unsigned long long)(__UINTPTR_TYPE__)gptr;
    u32x4 g0;
    g0[0] = 1u;                                   // count=1, user mode
    g0[1] = lds_off;                              // lds_addr
    g0[2] = (unsigned)ga;                         // global_addr lo
    g0[3] = ((unsigned)(ga >> 32) & 0x01ffffffu) | (2u << 30);  // hi + type=2
    u32x8 g1;
    g1[0] = (2u << 16) | (1u << 20) | (6u << 22) | (3u << 25);
    g1[1] = (128u << 16);                         // tensor_dim0 = 128
    g1[2] = (rows_avail & 0xffffu) << 16;         // tensor_dim1 lo16
    g1[3] = ((rows_avail >> 16) & 0xffffu) | (128u << 16);  // hi16, tile_dim0
    g1[4] = 128u;                                 // tile_dim1 = 128
    g1[5] = 128u;                                 // tensor_dim0_stride = 128
    g1[6] = 0u;
    g1[7] = 0u;
    u32x4 gz = {0u, 0u, 0u, 0u};
    asm volatile("tensor_load_to_lds %0, %1, %2, %3"
                 :: "s"(g0), "s"(g1), "s"(gz), "s"(gz) : "memory");
}

// ---------------------------------------------------------------------------
__global__ void zero_f32(float* __restrict__ p, int n) {
    for (int i = blockIdx.x * blockDim.x + threadIdx.x; i < n;
         i += gridDim.x * blockDim.x)
        p[i] = 0.0f;
}

// ---------------------------------------------------------------------------
// Edge kernel: acc[idnb[e]] += cb*cx * silu(x@Wx+bx) * silu((ca*rbf)@Wrbf+brbf)
// 256 threads = 8 waves; wave w owns rows [16w,16w+16), sweeps 8 col tiles:
// 4 WMMA (h, K=128) + 1 WMMA (gate, K=16 zero-padded to 32) per tile; both
// C tiles share the same lane layout, so the gate never touches LDS.
// ---------------------------------------------------------------------------
__global__ __launch_bounds__(256, 1)
void edge_gemm_gate_scatter(const float* __restrict__ x,
                            const float* __restrict__ rbf,
                            const int*   __restrict__ idnb,
                            const float* __restrict__ Wx,
                            const float* __restrict__ bx,
                            const float* __restrict__ Wrbf,
                            const float* __restrict__ brbf,
                            const float* __restrict__ coef_a,
                            const float* __restrict__ coef_b,
                            const float* __restrict__ coef_x,
                            float* __restrict__ acc,
                            int n_edges)
{
    extern __shared__ char smem[];
    float*    sXf  = (float*)smem;                  // [TE][XP]   x tile (TDM)
    _Float16* sWt  = (_Float16*)(sXf + TE * XP);    // [EMB][KP]  Wx^T
    _Float16* sWrT = sWt + EMB * KP;                // [EMB][RP]  Wrbf^T, K16..31=0
    float*    sRbf = (float*)(sWrT + EMB * RP);     // [TE][17]   ca*rbf
    float*    sB   = sRbf + TE * 17;                // bx
    float*    sBr  = sB + EMB;                      // brbf
    int*      sIdx = (int*)(sBr + EMB);             // [TE]

    const int tid  = threadIdx.x;
    const int e0   = blockIdx.x * TE;
    const int wave = tid >> 5;
    const int lane = tid & 31;

    // -- wave 0: kick off TDM DMA of the x tile ----------------------------
    if (wave == 0) {
        tdm_load_tile_f32(x + (size_t)e0 * EMB, 0u, (unsigned)(n_edges - e0));
    }

    // -- stage Wx^T (f16), Wrbf^T (f16, zero-padded K), biases, rbf, idx ---
    for (int i = tid; i < EMB * EMB; i += 256) {
        int k = i >> 7, n = i & 127;
        sWt[n * KP + k] = (_Float16)Wx[i];
    }
    for (int i = tid; i < EMB * 32; i += 256) {
        int n = i >> 5, k = i & 31;
        sWrT[n * RP + k] = (k < NRBF) ? (_Float16)Wrbf[k * EMB + n] : (_Float16)0.0f;
    }
    if (tid < EMB) { sB[tid] = bx[tid]; sBr[tid] = brbf[tid]; }

    const float ca = coef_a[0];
    for (int i = tid; i < TE * NRBF; i += 256) {
        int e = i >> 4, r = i & 15;
        int ge = e0 + e;
        sRbf[e * 17 + r] = (ge < n_edges) ? ca * rbf[(size_t)ge * NRBF + r] : 0.0f;
    }
    if (tid < TE) {
        int ge = e0 + tid;
        sIdx[tid] = (ge < n_edges) ? idnb[ge] : 0;
        int pe = ge + TE;                           // prefetch next tile
        if (pe < n_edges) __builtin_prefetch(&x[(size_t)pe * EMB], 0, 0);
    }

    if (wave == 0) __builtin_amdgcn_s_wait_tensorcnt(0);
    __syncthreads();

    // -- fragments ---------------------------------------------------------
    const int ml = lane & 15;   // A row / B,C column within 16x16 tile
    const int hi = lane >> 4;

    // A fragments for h (16x32 f16, ISA 7.12.2), from the f32 TDM tile
    const float* aRow = sXf + (wave * 16 + ml) * XP;
    v16h afrag[4];
    #pragma unroll
    for (int kc = 0; kc < 4; ++kc) {
        #pragma unroll
        for (int j = 0; j < 8; ++j) {
            int ka = kc * 32 + ((j < 4) ? (2 * j + 8 * hi)
                                        : (16 + 2 * (j - 4) + 8 * hi));
            afrag[kc][2 * j]     = (_Float16)aRow[ka];
            afrag[kc][2 * j + 1] = (_Float16)aRow[ka + 1];
        }
    }
    // A fragment for the gate (rbf row, K=0..15 valid, 16..31 zero)
    const float* rRow = sRbf + (wave * 16 + ml) * 17;
    v16h ar;
    #pragma unroll
    for (int j = 0; j < 8; ++j) {
        if (j < 4) {
            int ka = 2 * j + 8 * hi;                // < 16
            ar[2 * j]     = (_Float16)rRow[ka];
            ar[2 * j + 1] = (_Float16)rRow[ka + 1];
        } else {
            ar[2 * j]     = (_Float16)0.0f;         // K >= 16
            ar[2 * j + 1] = (_Float16)0.0f;
        }
    }

    const float gscale = coef_b[0] * coef_x[0];
    const bool  full   = (e0 + TE) <= n_edges;      // block-uniform

    for (int ct = 0; ct < 8; ++ct) {
        const int n = ct * 16 + ml;                 // output column
        const _Float16* bRow  = sWt  + n * KP;
        const _Float16* bRowR = sWrT + n * RP;

        v16h bfrag[4], bfr;
        #pragma unroll
        for (int kc = 0; kc < 4; ++kc) {
            #pragma unroll
            for (int j = 0; j < 8; ++j) {
                int kb = kc * 32 + ((j < 4) ? (2 * j + 8 * hi) : (16 + 2 * (j - 4) + 8 * hi));
                bfrag[kc][2 * j]     = bRow[kb];
                bfrag[kc][2 * j + 1] = bRow[kb + 1];
            }
        }
        #pragma unroll
        for (int j = 0; j < 8; ++j) {
            int kb = (j < 4) ? (2 * j + 8 * hi) : (16 + 2 * (j - 4) + 8 * hi);
            bfr[2 * j]     = bRowR[kb];
            bfr[2 * j + 1] = bRowR[kb + 1];
        }

        v8f c = {};
        #pragma unroll
        for (int kc = 0; kc < 4; ++kc)
            c = __builtin_amdgcn_wmma_f32_16x16x32_f16(
                    false, afrag[kc], false, bfrag[kc], (short)0, c, false, false);
        v8f cg = {};
        cg = __builtin_amdgcn_wmma_f32_16x16x32_f16(
                 false, ar, false, bfr, (short)0, cg, false, false);

        const float bias  = sB[n];
        const float biasr = sBr[n];
        if (full) {
            #pragma unroll
            for (int r = 0; r < 8; ++r) {
                int e = wave * 16 + r + 8 * hi;     // C row -> edge
                float h = silu_f(c[r]  + bias);
                float g = silu_f(cg[r] + biasr);
                atomicAdd(&acc[(size_t)sIdx[e] * EMB + n], gscale * h * g);
            }
        } else {
            #pragma unroll
            for (int r = 0; r < 8; ++r) {
                int e = wave * 16 + r + 8 * hi;
                float h = silu_f(c[r]  + bias);
                float g = silu_f(cg[r] + biasr);
                float m = (e0 + e < n_edges) ? 1.0f : 0.0f;   // cndmask, no branch
                atomicAdd(&acc[(size_t)sIdx[e] * EMB + n], m * gscale * h * g);
            }
        }
    }
}

// ---------------------------------------------------------------------------
// Atom-level dense layer: out = silu(in @ W + b). TDM-staged input tile,
// same WMMA tiling; block-uniform fast path when the whole tile is in range.
// ---------------------------------------------------------------------------
__global__ __launch_bounds__(256, 1)
void atom_gemm_silu(const float* __restrict__ in,
                    const float* __restrict__ W,
                    const float* __restrict__ b,
                    float* __restrict__ out,
                    int nrows)
{
    extern __shared__ char smem[];
    float*    sXf = (float*)smem;                 // [TE][XP]
    _Float16* sWt = (_Float16*)(sXf + TE * XP);   // [EMB][KP]
    float*    sB  = (float*)(sWt + EMB * KP);     // [EMB]

    const int tid  = threadIdx.x;
    const int r0   = blockIdx.x * TE;
    const int wave = tid >> 5;
    const int lane = tid & 31;

    if (wave == 0) {
        tdm_load_tile_f32(in + (size_t)r0 * EMB, 0u, (unsigned)(nrows - r0));
    }
    for (int i = tid; i < EMB * EMB; i += 256) {
        int k = i >> 7, n = i & 127;
        sWt[n * KP + k] = (_Float16)W[i];
    }
    if (tid < EMB) sB[tid] = b[tid];
    if (wave == 0) __builtin_amdgcn_s_wait_tensorcnt(0);
    __syncthreads();

    const int ml = lane & 15;
    const int hi = lane >> 4;

    const float* aRow = sXf + (wave * 16 + ml) * XP;
    v16h afrag[4];
    #pragma unroll
    for (int kc = 0; kc < 4; ++kc) {
        #pragma unroll
        for (int j = 0; j < 8; ++j) {
            int ka = kc * 32 + ((j < 4) ? (2 * j + 8 * hi)
                                        : (16 + 2 * (j - 4) + 8 * hi));
            afrag[kc][2 * j]     = (_Float16)aRow[ka];
            afrag[kc][2 * j + 1] = (_Float16)aRow[ka + 1];
        }
    }

    const bool full = (r0 + TE) <= nrows;         // block-uniform

    for (int ct = 0; ct < 8; ++ct) {
        const int n = ct * 16 + ml;
        const _Float16* bRow = sWt + n * KP;
        v16h bfrag[4];
        #pragma unroll
        for (int kc = 0; kc < 4; ++kc) {
            #pragma unroll
            for (int j = 0; j < 8; ++j) {
                int kb = kc * 32 + ((j < 4) ? (2 * j + 8 * hi) : (16 + 2 * (j - 4) + 8 * hi));
                bfrag[kc][2 * j]     = bRow[kb];
                bfrag[kc][2 * j + 1] = bRow[kb + 1];
            }
        }
        v8f c = {};
        #pragma unroll
        for (int kc = 0; kc < 4; ++kc)
            c = __builtin_amdgcn_wmma_f32_16x16x32_f16(
                    false, afrag[kc], false, bfrag[kc], (short)0, c, false, false);

        const float bias = sB[n];
        if (full) {
            #pragma unroll
            for (int r = 0; r < 8; ++r) {
                int e = wave * 16 + r + 8 * hi;
                out[(size_t)(r0 + e) * EMB + n] = silu_f(c[r] + bias);
            }
        } else {
            #pragma unroll
            for (int r = 0; r < 8; ++r) {
                int e = wave * 16 + r + 8 * hi;
                if (r0 + e < nrows)
                    out[(size_t)(r0 + e) * EMB + n] = silu_f(c[r] + bias);
            }
        }
    }
}

// ---------------------------------------------------------------------------
// Final projection: out[i] = coef * dot(a[i,:], Wf[:,0]); one wave per atom.
// ---------------------------------------------------------------------------
__global__ __launch_bounds__(256, 1)
void final_dot(const float* __restrict__ a,
               const float* __restrict__ Wf,
               const float* __restrict__ coef,
               float* __restrict__ out,
               int n)
{
    const int wave = threadIdx.x >> 5;
    const int lane = threadIdx.x & 31;
    const int atom = blockIdx.x * 8 + wave;
    if (atom >= n) return;

    const float4 av = *(const float4*)(a + (size_t)atom * EMB + lane * 4);
    const float4 wv = *(const float4*)(Wf + lane * 4);
    float s = av.x * wv.x + av.y * wv.y + av.z * wv.z + av.w * wv.w;
    #pragma unroll
    for (int o = 16; o > 0; o >>= 1) s += __shfl_xor(s, o, 32);
    if (lane == 0) out[atom] = s * coef[0];
}

// ---------------------------------------------------------------------------
extern "C" void kernel_launch(void* const* d_in, const int* in_sizes, int n_in,
                              void* d_out, int out_size, void* d_ws, size_t ws_size,
                              hipStream_t stream)
{
    const float* x     = (const float*)d_in[0];
    const float* rbf   = (const float*)d_in[1];
    const int*   idnb  = (const int*)  d_in[2];
    /* d_in[3] = n_atoms device scalar; host uses out_size instead */
    const float* Wx    = (const float*)d_in[4];
    const float* bx    = (const float*)d_in[5];
    const float* Wrbf  = (const float*)d_in[6];
    const float* brbf  = (const float*)d_in[7];
    const float* W1    = (const float*)d_in[8];
    const float* b1    = (const float*)d_in[9];
    const float* W2    = (const float*)d_in[10];
    const float* b2    = (const float*)d_in[11];
    const float* W3    = (const float*)d_in[12];
    const float* b3    = (const float*)d_in[13];
    const float* Wf    = (const float*)d_in[14];
    const float* ca    = (const float*)d_in[15];
    const float* cb    = (const float*)d_in[16];
    const float* cx    = (const float*)d_in[17];
    const float* cf    = (const float*)d_in[18];
    float* out = (float*)d_out;

    const int n_edges = in_sizes[0] / EMB;   // 1,000,000
    const int n_atoms = out_size;            // 50,000

    float* acc = (float*)d_ws;                   // [n_atoms][128]
    float* tmp = acc + (size_t)n_atoms * EMB;    // [n_atoms][128]

    zero_f32<<<2048, 256, 0, stream>>>(acc, n_atoms * EMB);

    const int eblocks = (n_edges + TE - 1) / TE;
    edge_gemm_gate_scatter<<<eblocks, 256, SMEM_EDGE, stream>>>(
        x, rbf, idnb, Wx, bx, Wrbf, brbf, ca, cb, cx, acc, n_edges);

    const int ablocks = (n_atoms + TE - 1) / TE;
    atom_gemm_silu<<<ablocks, 256, SMEM_ATOM, stream>>>(acc, W1, b1, tmp, n_atoms);
    atom_gemm_silu<<<ablocks, 256, SMEM_ATOM, stream>>>(tmp, W2, b2, acc, n_atoms);
    atom_gemm_silu<<<ablocks, 256, SMEM_ATOM, stream>>>(acc, W3, b3, tmp, n_atoms);

    final_dot<<<(n_atoms + 7) / 8, 256, 0, stream>>>(tmp, Wf, cf, out, n_atoms);
}
